// Generate_36936718745868
// MI455X (gfx1250) — matrix operand order repaired and checked by reading, not compiled, not run on hardware
//
#include <hip/hip_runtime.h>
#include <hip/hip_bf16.h>

// ---- problem constants (from reference setup_inputs) ----
#define V_     10000
#define E_     512
#define H_     1024
#define G4_    4096      // 4*H
#define B_     32
#define BEAM_  4
#define NMAX   128       // B*BEAM
#define T_     16        // "length" in setup_inputs (device scalar; fixed)
#define PAD_ID 0
#define EOS_ID 1
#define LPF_   0.7f

#define CT_G   4         // column tiles per wave, gates GEMM   (256 = 64*4)
#define CT_P   5         // column tiles per wave, proj GEMM    (625 = 125*5)

typedef __attribute__((ext_vector_type(16))) __bf16 bf16x16;
typedef __attribute__((ext_vector_type(8)))  __bf16 bf16x8;
typedef __attribute__((ext_vector_type(8)))  float  f32x8;

__device__ __forceinline__ __bf16 f2bf(float f) {
  unsigned u = __float_as_uint(f);
  u += 0x7FFFu + ((u >> 16) & 1u);               // round-to-nearest-even
  unsigned short s = (unsigned short)(u >> 16);
  return __builtin_bit_cast(__bf16, s);
}

// ---------------------------------------------------------------------------
// Weight preparation: f32 [K x N] row-major -> bf16 WMMA B-fragment packing.
// Fragment for (n-tile, k-tile): 32 lanes x 16 bf16, laid out per ISA 7.12.2:
//   B (32x16, 16-bit): lanes 0-15 hold K={0..7,16..23}, lanes 16-31 hold K={8..15,24..31}; N=lane%16.
// Packed so the GEMM inner loop reads one contiguous 32-byte run (2x b128) per lane per B frag.
// ---------------------------------------------------------------------------
__global__ void pack_b_frags(const float* __restrict__ W, __bf16* __restrict__ out,
                             int K, int N) {
  int t = blockIdx.x * blockDim.x + threadIdx.x;
  int total = K * N;
  if (t >= total) return;
  int e    = t & 15;
  int lane = (t >> 4) & 31;
  int frag = t >> 9;
  int KT   = K >> 5;
  int kt   = frag % KT;
  int nt   = frag / KT;
  int col  = nt * 16 + (lane & 15);
  int k    = kt * 32 + (e & 7) + ((e & 8) ? 16 : 0) + ((lane & 16) ? 8 : 0);
  out[t] = f2bf(W[(size_t)k * N + col]);
}

__global__ void cvt_f32_bf16(const float* __restrict__ in, __bf16* __restrict__ out, int n) {
  int i = blockIdx.x * blockDim.x + threadIdx.x;
  if (i < n) out[i] = f2bf(in[i]);
}

// ---------------------------------------------------------------------------
// One-time concept projections: [32 x 10000] @ [10000 x 4096].
// Pure bandwidth (328 MB streamed once at 23.3 TB/s) -> plain f32 FMA.
// ---------------------------------------------------------------------------
__global__ void concept_gemm(const float* __restrict__ C, const float* __restrict__ W,
                             float* __restrict__ out) {
  int col = blockIdx.x * blockDim.x + threadIdx.x;   // 0..4095 (coalesced on W)
  int b   = blockIdx.y;                              // 0..31
  const float* crow = C + (size_t)b * V_;
  float acc = 0.f;
  for (int k = 0; k < V_; ++k)
    acc = __fmaf_rn(crow[k], W[(size_t)k * G4_ + col], acc);
  out[(size_t)b * G4_ + col] = acc;
}

// ---------------------------------------------------------------------------
// WMMA fragment loaders (wave32).
// A (16x32, 16-bit, ISA 7.12.2): lane holds row M=lane%16; lanes<16 get
// K = {0..7, 16..23}, lanes>=16 get K = {8..15, 24..31}: two b128 loads.
// ---------------------------------------------------------------------------
__device__ __forceinline__ bf16x16 load_a_frag(const __bf16* __restrict__ act,
                                               int ldk, int m0, int k0, int lane) {
  int row  = m0 + (lane & 15);
  int half = (lane & 16) ? 8 : 0;
  const __bf16* p = act + (size_t)row * ldk + k0 + half;
  bf16x8 lo = *reinterpret_cast<const bf16x8*>(p);
  bf16x8 hi = *reinterpret_cast<const bf16x8*>(p + 16);
  return __builtin_shufflevector(lo, hi, 0,1,2,3,4,5,6,7,8,9,10,11,12,13,14,15);
}

__device__ __forceinline__ bf16x16 load_b_frag(const __bf16* __restrict__ packed,
                                               int KT, int nt, int kt, int lane) {
  return *reinterpret_cast<const bf16x16*>(packed + (((size_t)nt * KT + kt) * 32 + lane) * 16);
}

// ---------------------------------------------------------------------------
// Software-pipelined fragment MAC loop: while the WMMAs of k-step kt execute,
// the A fragment + CT B fragments of kt+1 are already in flight in their own
// registers -> each WMMA waits only on loads issued one full k-step earlier.
// ---------------------------------------------------------------------------
template <int KT, int CT>
__device__ __forceinline__ void wmma_pipeline(const __bf16* __restrict__ act, int ldk,
                                              const __bf16* __restrict__ Wpk,
                                              int nt0, int m0, int lane, f32x8* acc) {
  bf16x16 a = load_a_frag(act, ldk, m0, 0, lane);
  bf16x16 bt[CT];
#pragma unroll
  for (int c = 0; c < CT; ++c) bt[c] = load_b_frag(Wpk, KT, nt0 + c, 0, lane);

  for (int kt = 0; kt < KT - 1; ++kt) {
    bf16x16 a_n = load_a_frag(act, ldk, m0, (kt + 1) * 32, lane);
    bf16x16 bt_n[CT];
#pragma unroll
    for (int c = 0; c < CT; ++c) bt_n[c] = load_b_frag(Wpk, KT, nt0 + c, kt + 1, lane);
#pragma unroll
    for (int c = 0; c < CT; ++c)
      acc[c] = __builtin_amdgcn_wmma_f32_16x16x32_bf16(false, a, false, bt[c], (short)0, acc[c], false, false);
    a = a_n;
#pragma unroll
    for (int c = 0; c < CT; ++c) bt[c] = bt_n[c];
  }
#pragma unroll
  for (int c = 0; c < CT; ++c)
    acc[c] = __builtin_amdgcn_wmma_f32_16x16x32_bf16(false, a, false, bt[c], (short)0, acc[c], false, false);
}

// ---------------------------------------------------------------------------
// Gate GEMM: g = sigmoid(CI*(x@Wi) + CS*(h@Ws)), output [N x 4096].
// One wave per 16 x (4*16) output strip, double-buffered fragment pipeline.
// ---------------------------------------------------------------------------
__global__ void gemm_gates(const __bf16* __restrict__ x_bf, const __bf16* __restrict__ h_bf,
                           const __bf16* __restrict__ Wi_p, const __bf16* __restrict__ Ws_p,
                           const float* __restrict__ CI, const float* __restrict__ CS,
                           float* __restrict__ g, int beam_div) {
  int lane = threadIdx.x;          // full wave active -> EXEC all ones (WMMA req)
  int nt0  = blockIdx.x * CT_G;    // 0..255 in steps of 4
  int m0   = blockIdx.y * 16;
  f32x8 acc1[CT_G], acc2[CT_G];
#pragma unroll
  for (int c = 0; c < CT_G; ++c) {
    acc1[c] = (f32x8){0.f,0.f,0.f,0.f,0.f,0.f,0.f,0.f};
    acc2[c] = (f32x8){0.f,0.f,0.f,0.f,0.f,0.f,0.f,0.f};
  }
  wmma_pipeline<E_ / 32, CT_G>(x_bf, E_, Wi_p, nt0, m0, lane, acc1);
  wmma_pipeline<H_ / 32, CT_G>(h_bf, H_, Ws_p, nt0, m0, lane, acc2);

  // C/D layout: element v -> M = v + (lane>=16 ? 8 : 0), N = lane%16
  int mbase = m0 + ((lane & 16) ? 8 : 0);
#pragma unroll
  for (int c = 0; c < CT_G; ++c) {
    int nc = (nt0 + c) * 16 + (lane & 15);
#pragma unroll
    for (int v = 0; v < 8; ++v) {
      int row = mbase + v;
      size_t cb = (size_t)(row / beam_div) * G4_ + nc;   // CI/CS repeated per beam
      float pre = CI[cb] * acc1[c][v] + CS[cb] * acc2[c][v];
      g[(size_t)row * G4_ + nc] = 1.f / (1.f + __expf(-pre));
    }
  }
}

// ---------------------------------------------------------------------------
// Projection GEMM: logits = h2 @ W_proj + b_proj, output [N x 10000].
// 5 column tiles per wave (625 = 125*5), double-buffered fragment pipeline.
// ---------------------------------------------------------------------------
__global__ void gemm_logits(const __bf16* __restrict__ h_bf, const __bf16* __restrict__ Wp_p,
                            const float* __restrict__ bias, float* __restrict__ logits) {
  int lane = threadIdx.x;
  int nt0  = blockIdx.x * CT_P;    // 0..624 in steps of 5
  int m0   = blockIdx.y * 16;
  f32x8 acc[CT_P];
#pragma unroll
  for (int c = 0; c < CT_P; ++c)
    acc[c] = (f32x8){0.f,0.f,0.f,0.f,0.f,0.f,0.f,0.f};
  wmma_pipeline<H_ / 32, CT_P>(h_bf, H_, Wp_p, nt0, m0, lane, acc);

  int mbase = m0 + ((lane & 16) ? 8 : 0);
#pragma unroll
  for (int c = 0; c < CT_P; ++c) {
    int nc = (nt0 + c) * 16 + (lane & 15);
    float bv = bias[nc];
#pragma unroll
    for (int v = 0; v < 8; ++v)
      logits[(size_t)(mbase + v) * V_ + nc] = acc[c][v] + bv;
  }
}

// ---------------------------------------------------------------------------
// LSTM pointwise: c2 = i*cc + f*c ; h2 = o*tanh(c2)
// ---------------------------------------------------------------------------
__global__ void lstm_pointwise(const float* __restrict__ g, const float* __restrict__ c_prev,
                               float* __restrict__ c_out, float* __restrict__ h_out, int Nrows) {
  int idx = blockIdx.x * blockDim.x + threadIdx.x;
  if (idx >= Nrows * H_) return;
  int r = idx >> 10, j = idx & (H_ - 1);
  const float* gr = g + (size_t)r * G4_;
  float i  = gr[j];
  float f  = gr[H_ + j];
  float o  = gr[2 * H_ + j];
  float cc = gr[3 * H_ + j];
  float c2 = i * cc + f * c_prev[idx];
  c_out[idx] = c2;
  h_out[idx] = o * tanhf(c2);
}

// ---------------------------------------------------------------------------
// softmax over V, then beam scores (length penalty); logits overwritten by score
// ---------------------------------------------------------------------------
__global__ void softmax_score(float* __restrict__ logits, float* __restrict__ wp2,
                              const int* __restrict__ stops, const float* __restrict__ prob,
                              const int* __restrict__ wl) {
  int r = blockIdx.x, tid = threadIdx.x;
  __shared__ float red[256];
  float* L = logits + (size_t)r * V_;
  float mx = -1e30f;
  for (int v = tid; v < V_; v += 256) mx = fmaxf(mx, L[v]);
  red[tid] = mx; __syncthreads();
  for (int s = 128; s > 0; s >>= 1) { if (tid < s) red[tid] = fmaxf(red[tid], red[tid + s]); __syncthreads(); }
  mx = red[0]; __syncthreads();
  float sum = 0.f;
  for (int v = tid; v < V_; v += 256) sum += __expf(L[v] - mx);
  red[tid] = sum; __syncthreads();
  for (int s = 128; s > 0; s >>= 1) { if (tid < s) red[tid] += red[tid + s]; __syncthreads(); }
  float inv = 1.f / red[0];
  int   sf  = stops[r];
  float pr  = prob[r];
  int   wlr = wl[r];
  const float inv6p = 0.2852951f;                 // 6^-0.7
  for (int v = tid; v < V_; v += 256) {
    float p  = __expf(L[v] - mx) * inv;
    float w2 = pr * (sf ? ((v == PAD_ID) ? 1.f : 0.f) : p);
    wp2[(size_t)r * V_ + v] = w2;
    int   unst = (v > EOS_ID) ? 1 : 0;
    float lp   = __powf((float)(wlr + unst * (1 - sf) + 5), LPF_) * inv6p;
    float w2c  = fminf(fmaxf(w2, 1e-20f), 1.0f);
    L[v] = __logf(w2c) / lp;                      // score in place
  }
}

// ---------------------------------------------------------------------------
// per-batch top-4 over beam_in*V contiguous scores
// ---------------------------------------------------------------------------
__global__ void topk4(const float* __restrict__ score, int* __restrict__ topk_idx, int beam_in) {
  int b = blockIdx.x, tid = threadIdx.x;
  int M = beam_in * V_;
  const float* S = score + (size_t)b * beam_in * V_;
  float bv[4] = {-1e30f, -1e30f, -1e30f, -1e30f};
  int   bi[4] = {0, 0, 0, 0};
  for (int j = tid; j < M; j += 256) {
    float v = S[j];
    if (v > bv[3]) {
      int p = 3;
      while (p > 0 && v > bv[p - 1]) { bv[p] = bv[p - 1]; bi[p] = bi[p - 1]; --p; }
      bv[p] = v; bi[p] = j;
    }
  }
  __shared__ float sv[256][4];
  __shared__ int   si[256][4];
#pragma unroll
  for (int q = 0; q < 4; ++q) { sv[tid][q] = bv[q]; si[tid][q] = bi[q]; }
  __syncthreads();
  for (int s = 128; s > 0; s >>= 1) {
    if (tid < s) {
#pragma unroll
      for (int q = 0; q < 4; ++q) {
        float v = sv[tid + s][q]; int ix = si[tid + s][q];
        if (v > sv[tid][3]) {
          int p = 3;
          while (p > 0 && v > sv[tid][p - 1]) { sv[tid][p] = sv[tid][p - 1]; si[tid][p] = si[tid][p - 1]; --p; }
          sv[tid][p] = v; si[tid][p] = ix;
        }
      }
    }
    __syncthreads();
  }
  if (tid < 4) topk_idx[b * 4 + tid] = si[0][tid];
}

// ---------------------------------------------------------------------------
// beam select/update: decode winners, gather h/c along prev, carry state
// ---------------------------------------------------------------------------
__global__ void beam_update(const int* __restrict__ topk_idx, const float* __restrict__ wp2,
                            const float* __restrict__ h2, const float* __restrict__ c2,
                            const int* __restrict__ wl_in,
                            __bf16* __restrict__ h_bf_out, float* __restrict__ c_out,
                            int* __restrict__ word_next, int* __restrict__ stops_next,
                            float* __restrict__ prob_next, int* __restrict__ wl_next,
                            int* __restrict__ words_all, int* __restrict__ prevs_all,
                            int beam_in, int t) {
  int r  = blockIdx.x;                 // 0..127 new beams
  int b  = r >> 2, j = r & 3;
  int bw = topk_idx[b * 4 + j];
  int prev = b * beam_in + bw / V_;
  int nw   = bw % V_;
  int st2  = (nw <= EOS_ID) ? 1 : 0;
  if (threadIdx.x == 0) {
    prob_next[r]  = wp2[(size_t)prev * V_ + nw];
    wl_next[r]    = wl_in[prev] + (1 - st2);
    stops_next[r] = st2;
    word_next[r]  = nw;
    words_all[t * NMAX + r] = nw;
    prevs_all[t * NMAX + r] = prev;
  }
  for (int i = threadIdx.x; i < H_; i += blockDim.x) {
    h_bf_out[(size_t)r * H_ + i] = f2bf(h2[(size_t)prev * H_ + i]);
    c_out[(size_t)r * H_ + i]    = c2[(size_t)prev * H_ + i];
  }
}

__global__ void gather_embed(const int* __restrict__ word, const __bf16* __restrict__ emb,
                             __bf16* __restrict__ x_bf) {
  int r = blockIdx.x;
  int w = word[r];
  for (int i = threadIdx.x; i < E_; i += blockDim.x)
    x_bf[(size_t)r * E_ + i] = emb[(size_t)w * E_ + i];
}

__global__ void init_state(int* __restrict__ stops, float* __restrict__ prob, int* __restrict__ wl) {
  int i = threadIdx.x;                 // 128 threads
  stops[i] = 0; prob[i] = 1.f; wl[i] = 0;
}

__global__ void traceback(const int* __restrict__ words_all, const int* __restrict__ prevs_all,
                          float* __restrict__ out) {
  int b = threadIdx.x;
  if (b >= B_) return;
  int idx = b * BEAM_;
  for (int t = T_ - 1; t >= 0; --t) {
    out[b * T_ + t] = (float)words_all[t * NMAX + idx];
    idx = prevs_all[t * NMAX + idx];
  }
}

// ---------------------------------------------------------------------------
extern "C" void kernel_launch(void* const* d_in, const int* in_sizes, int n_in,
                              void* d_out, int out_size, void* d_ws, size_t ws_size,
                              hipStream_t stream) {
  (void)in_sizes; (void)n_in; (void)out_size; (void)ws_size;
  // setup_inputs order:
  // 0 length (device scalar, == 16), 1 word, 2 concept, 3 word_emb, 4 W_input,
  // 5 W_state, 6 W_ci, 7 W_cs, 8 W_proj, 9 b_proj, 10 h0, 11 c0
  const int*   word_in  = (const int*)  d_in[1];
  const float* concept  = (const float*)d_in[2];
  const float* word_emb = (const float*)d_in[3];
  const float* W_input  = (const float*)d_in[4];
  const float* W_state  = (const float*)d_in[5];
  const float* W_ci     = (const float*)d_in[6];
  const float* W_cs     = (const float*)d_in[7];
  const float* W_proj   = (const float*)d_in[8];
  const float* b_proj   = (const float*)d_in[9];
  const float* h0       = (const float*)d_in[10];
  const float* c0       = (const float*)d_in[11];

  char* ws = (char*)d_ws;
  size_t off = 0;
  auto alloc = [&](size_t bytes) -> void* {
    void* p = ws + off; off += (bytes + 255) & ~(size_t)255; return p;
  };
  __bf16* Wi_p   = (__bf16*)alloc((size_t)E_ * G4_ * 2);
  __bf16* Ws_p   = (__bf16*)alloc((size_t)H_ * G4_ * 2);
  __bf16* Wp_p   = (__bf16*)alloc((size_t)H_ * V_ * 2);
  __bf16* embBf  = (__bf16*)alloc((size_t)V_ * E_ * 2);
  float*  CI_B   = (float*) alloc((size_t)B_ * G4_ * 4);
  float*  CS_B   = (float*) alloc((size_t)B_ * G4_ * 4);
  __bf16* hBf    = (__bf16*)alloc((size_t)NMAX * H_ * 2);
  __bf16* xBf    = (__bf16*)alloc((size_t)NMAX * E_ * 2);
  float*  hCell  = (float*) alloc((size_t)NMAX * H_ * 4);
  float*  cCell  = (float*) alloc((size_t)NMAX * H_ * 4);
  float*  cCarry = (float*) alloc((size_t)NMAX * H_ * 4);
  float*  gbuf   = (float*) alloc((size_t)NMAX * G4_ * 4);
  float*  logits = (float*) alloc((size_t)NMAX * V_ * 4);   // reused as score
  float*  wp2    = (float*) alloc((size_t)NMAX * V_ * 4);
  int*    topk   = (int*)   alloc(B_ * BEAM_ * 4);
  int*    wordC  = (int*)   alloc(NMAX * 4);
  int*    stopsA = (int*)   alloc(NMAX * 4);
  int*    stopsB = (int*)   alloc(NMAX * 4);
  float*  probA  = (float*) alloc(NMAX * 4);
  float*  probB  = (float*) alloc(NMAX * 4);
  int*    wlA    = (int*)   alloc(NMAX * 4);
  int*    wlB    = (int*)   alloc(NMAX * 4);
  int*    wordsAll = (int*) alloc((size_t)T_ * NMAX * 4);
  int*    prevsAll = (int*) alloc((size_t)T_ * NMAX * 4);

  // ---- per-call weight prep (bf16 + WMMA-fragment packing; L2-resident) ----
  pack_b_frags<<<(E_ * G4_ + 255) / 256, 256, 0, stream>>>(W_input, Wi_p, E_, G4_);
  pack_b_frags<<<(H_ * G4_ + 255) / 256, 256, 0, stream>>>(W_state, Ws_p, H_, G4_);
  pack_b_frags<<<(H_ * V_  + 255) / 256, 256, 0, stream>>>(W_proj,  Wp_p, H_, V_);
  cvt_f32_bf16<<<(V_ * E_ + 255) / 256, 256, 0, stream>>>(word_emb, embBf, V_ * E_);
  concept_gemm<<<dim3(G4_ / 256, B_), 256, 0, stream>>>(concept, W_ci, CI_B);
  concept_gemm<<<dim3(G4_ / 256, B_), 256, 0, stream>>>(concept, W_cs, CS_B);
  init_state<<<1, NMAX, 0, stream>>>(stopsA, probA, wlA);
  cvt_f32_bf16<<<(B_ * H_ + 255) / 256, 256, 0, stream>>>(h0, hBf, B_ * H_);

  int*   stops_cur = stopsA; int*   stops_nxt = stopsB;
  float* prob_cur  = probA;  float* prob_nxt  = probB;
  int*   wl_cur    = wlA;    int*   wl_nxt    = wlB;

  for (int t = 0; t < T_; ++t) {
    const int N        = (t == 0) ? B_ : NMAX;
    const int beam_in  = (t == 0) ? 1  : BEAM_;
    const int beam_div = (t == 0) ? 1  : BEAM_;
    const int*   wsrc  = (t == 0) ? word_in : wordC;
    const float* cprev = (t == 0) ? c0 : cCarry;

    gather_embed<<<N, 256, 0, stream>>>(wsrc, embBf, xBf);
    gemm_gates<<<dim3(G4_ / 16 / CT_G, N / 16), 32, 0, stream>>>(xBf, hBf, Wi_p, Ws_p,
                                                                 CI_B, CS_B, gbuf, beam_div);
    lstm_pointwise<<<(N * H_ + 255) / 256, 256, 0, stream>>>(gbuf, cprev, cCell, hCell, N);
    cvt_f32_bf16<<<(N * H_ + 255) / 256, 256, 0, stream>>>(hCell, hBf, N * H_);
    gemm_logits<<<dim3(V_ / 16 / CT_P, N / 16), 32, 0, stream>>>(hBf, Wp_p, b_proj, logits);
    softmax_score<<<N, 256, 0, stream>>>(logits, wp2, stops_cur, prob_cur, wl_cur);
    topk4<<<B_, 256, 0, stream>>>(logits, topk, beam_in);
    beam_update<<<NMAX, 256, 0, stream>>>(topk, wp2, hCell, cCell, wl_cur,
                                          hBf, cCarry, wordC, stops_nxt, prob_nxt, wl_nxt,
                                          wordsAll, prevsAll, beam_in, t);
    { int* ti = stops_cur; stops_cur = stops_nxt; stops_nxt = ti; }
    { float* tf = prob_cur; prob_cur = prob_nxt; prob_nxt = tf; }
    { int* ti = wl_cur; wl_cur = wl_nxt; wl_nxt = ti; }
  }

  traceback<<<1, 32, 0, stream>>>(wordsAll, prevsAll, (float*)d_out);
}
